// RNN_80015240724707
// MI455X (gfx1250) — matrix operand order repaired and checked by reading, not compiled, not run on hardware
//
#include <hip/hip_runtime.h>
#include <hip/hip_bf16.h>
#include <stdint.h>

// ---------------------------------------------------------------------------
// Model dims
// ---------------------------------------------------------------------------
#define TSEQ    256
#define BATCH   8
#define MROWS   2048            // T*B, time-major rows (row = t*8 + b)
#define DEMB    512
#define HSTRIDE 1024            // row stride of h-state buffers in global
#define HLDS    1032            // padded LDS row stride (bank-conflict-free A rows)
#define GHLD    104             // padded LDS stride for gh staging tile

#define AS1 __attribute__((address_space(1)))
#define AS3 __attribute__((address_space(3)))

#if defined(__gfx1250__) && __has_builtin(__builtin_amdgcn_global_load_async_to_lds_b128)
#define ASYNC_LDS 1
#else
#define ASYNC_LDS 0
#endif

typedef __bf16  bf16x16 __attribute__((ext_vector_type(16)));
typedef float   f32x8   __attribute__((ext_vector_type(8)));
typedef unsigned int u32x4 __attribute__((ext_vector_type(4)));
typedef int     v4i     __attribute__((vector_size(16)));   // builtin's pointee type

union AFrag { bf16x16 v; u32x4 q[2]; };
union CFrag { f32x8  v; float f[8]; };

__device__ __forceinline__ unsigned short f2bf(float f) {
  union { float f; unsigned u; } v; v.f = f;
  unsigned r = v.u + 0x7FFFu + ((v.u >> 16) & 1u);   // RNE
  return (unsigned short)(r >> 16);
}

// ---------------------------------------------------------------------------
// Device-scope sense barrier (per-stack counters; 32 blocks/stack resident)
// ---------------------------------------------------------------------------
__device__ __forceinline__ void grid_barrier(int* cnt, int* gen, int nblk, int expect) {
  __syncthreads();
  if (threadIdx.x == 0) {
    __threadfence();
    int prev = __hip_atomic_fetch_add(cnt, 1, __ATOMIC_ACQ_REL, __HIP_MEMORY_SCOPE_AGENT);
    if (prev == nblk - 1) {
      __hip_atomic_store(cnt, 0, __ATOMIC_RELAXED, __HIP_MEMORY_SCOPE_AGENT);
      __hip_atomic_fetch_add(gen, 1, __ATOMIC_RELEASE, __HIP_MEMORY_SCOPE_AGENT);
    } else {
      while (__hip_atomic_load(gen, __ATOMIC_ACQUIRE, __HIP_MEMORY_SCOPE_AGENT) < expect)
        __builtin_amdgcn_s_sleep(2);
    }
  }
  __syncthreads();
}

// ---------------------------------------------------------------------------
// Misc prep kernels
// ---------------------------------------------------------------------------
__global__ void init_sync(int* sync) { if (threadIdx.x < 4) sync[threadIdx.x] = 0; }

__global__ void cvt_f32_bf16(const float* __restrict__ src, unsigned short* __restrict__ dst, int n) {
  int i = blockIdx.x * blockDim.x + threadIdx.x;
  int stride = gridDim.x * blockDim.x;
  for (; i < n; i += stride) dst[i] = f2bf(src[i]);
}

// x[t,b,:] = bf16(E[tokens[b,t]])  -> X0 row-major [2048, 512]
__global__ void gather_embed(const int* __restrict__ tokens, const float* __restrict__ E,
                             unsigned short* __restrict__ X0) {
  const int row = blockIdx.x;          // t*8 + b
  const int t = row >> 3, b = row & 7;
  const int tok = tokens[b * TSEQ + t];
  const float* e = E + (size_t)tok * DEMB;
  for (int c = threadIdx.x; c < DEMB; c += blockDim.x)
    X0[(size_t)row * DEMB + c] = f2bf(e[c]);
}

// ---------------------------------------------------------------------------
// WMMA bf16 GEMM:  out[M=2048,N] = A[2048,K] * W[N,K]^T + bias
// Register-blocked: each wave computes 16(M) x 64(N) = 4 accumulators sharing
// one A fragment; 8 waves/block -> 16 x 512 per block.
// A frag: lane L (row mBase+(L&15)) loads 16 contiguous bf16 at kk+(L>>4)*16
// B frag: lane L (W row nTile+(L&15)) loads 16 contiguous bf16 at kk+(L>>4)*16
// (contiguous k = 16*(L>>4) + s order, identical for A and B; not the ISA table order)
// ---------------------------------------------------------------------------
__global__ void wmma_gemm_bias(const unsigned short* __restrict__ A,
                               const unsigned short* __restrict__ W,
                               const float* __restrict__ bias,
                               float* __restrict__ out,
                               int N, int K) {
  const int lane  = threadIdx.x & 31;
  const int wave  = threadIdx.x >> 5;
  const int nBase = blockIdx.x * 512 + wave * 64;
  if (nBase >= N) return;                       // wave-uniform: EXEC stays all-1
  const int mBase = blockIdx.y * 16;
  const int arow  = mBase + (lane & 15);
  const unsigned short* aPtr = A + (size_t)arow * K + ((lane >> 4) * 16);

  const unsigned short* wPtr[4];
#pragma unroll
  for (int tt = 0; tt < 4; ++tt) {
    int wr = nBase + tt * 16 + (lane & 15);
    if (wr >= N) wr = N - 1;                     // clamp tail reads
    wPtr[tt] = W + (size_t)wr * K + ((lane >> 4) * 16);
  }

  f32x8 acc[4] = {};
  for (int kk = 0; kk < K; kk += 32) {
    AFrag a;
    a.q[0] = *(const u32x4*)(aPtr + kk);
    a.q[1] = *(const u32x4*)(aPtr + kk + 8);
    __builtin_prefetch(wPtr[0] + kk + 512, 0, 0);        // global_prefetch_b8
#pragma unroll
    for (int tt = 0; tt < 4; ++tt) {
      AFrag b;
      b.q[0] = *(const u32x4*)(wPtr[tt] + kk);
      b.q[1] = *(const u32x4*)(wPtr[tt] + kk + 8);
      acc[tt] = __builtin_amdgcn_wmma_f32_16x16x32_bf16(false, a.v, false, b.v,
                                                        (short)0, acc[tt], false, false);
    }
  }

  const int mo = (lane >> 4) * 8;                        // C layout: M = r + 8*(lane>=16)
#pragma unroll
  for (int tt = 0; tt < 4; ++tt) {
    const int n = nBase + tt * 16 + (lane & 15);
    if (n < N) {
      const float bb = bias ? bias[n] : 0.0f;
      CFrag c; c.v = acc[tt];
      for (int r = 0; r < 8; ++r)
        out[(size_t)(mBase + mo + r) * N + n] = c.f[r] + bb;
    }
  }
}

// ---------------------------------------------------------------------------
// GRU recurrence: sequential over T. 32 WGs per stack each own ncols=H/32
// h-columns; per step: stage h (bf16, ping-pong in global/L2) into LDS via
// GLOBAL_LOAD_ASYNC_TO_LDS (ASYNCcnt DMA), compute gh slice = h @ Whh_slice^T
// with WMMA (A frags from LDS, bank-conflict-free padded stride), fuse gates,
// device-barrier, next step. gridDim = (32, 2 stacks), blockDim = 256.
// ---------------------------------------------------------------------------
__global__ void gru_recurrence(const float* __restrict__ Gi_g, const float* __restrict__ Gi_s,
                               const unsigned short* __restrict__ Whh_g,
                               const unsigned short* __restrict__ Whh_s,
                               const float* __restrict__ bhh_g, const float* __restrict__ bhh_s,
                               unsigned short* __restrict__ Xout_g, unsigned short* __restrict__ Xout_s,
                               float* __restrict__ hf32, unsigned short* __restrict__ hbf,
                               int* __restrict__ sync, int Hl) {
  const int stack = blockIdx.y;
  const float* Gi            = stack ? Gi_s   : Gi_g;
  const unsigned short* Whh  = stack ? Whh_s  : Whh_g;
  const float* bhh           = stack ? bhh_s  : bhh_g;
  unsigned short* Xout       = stack ? Xout_s : Xout_g;
  float* hS          = hf32 + (size_t)stack * 8 * HSTRIDE;
  unsigned short* hB = hbf  + (size_t)stack * 2 * 16 * HSTRIDE;   // ping-pong [2][16][HSTRIDE]
  int* cnt = sync + stack * 2;
  int* gen = sync + stack * 2 + 1;

  const int nWG    = gridDim.x;          // 32
  const int ncols  = Hl / nWG;           // 32 (H=1024) or 16 (H=512)
  const int cs     = blockIdx.x * ncols;
  const int tpg    = ncols / 16;         // 16x16 N-tiles per gate
  const int ntiles = 3 * tpg;            // 6 or 3
  const int tid = threadIdx.x, lane = tid & 31, wave = tid >> 5;

  __shared__ __align__(16) unsigned short h_lds[16 * HLDS];  // staged h, padded rows
  __shared__ float ghlds[16 * GHLD];                         // gh[m][tile*16+n]
  __shared__ int s_gen0;
  if (tid == 0) s_gen0 = __hip_atomic_load(gen, __ATOMIC_RELAXED, __HIP_MEMORY_SCOPE_AGENT);

  // zero h state (rows 8..15 of bf16 h stay zero forever -> padded WMMA A)
  for (int i = blockIdx.x * blockDim.x + tid; i < 8 * HSTRIDE;      i += nWG * blockDim.x) hS[i] = 0.0f;
  for (int i = blockIdx.x * blockDim.x + tid; i < 2 * 16 * HSTRIDE; i += nWG * blockDim.x) hB[i] = 0;
  grid_barrier(cnt, gen, nWG, s_gen0 + 1);

  for (int t = 0; t < TSEQ; ++t) {
    const unsigned short* hRead = hB + (t & 1) * 16 * HSTRIDE;
    unsigned short* hWrite      = hB + ((t + 1) & 1) * 16 * HSTRIDE;

    // ---- stage h[16, Hl] into LDS (async DMA when available) ----
    {
      const int chunksPerRow = HSTRIDE / 8;                // 16-byte chunks
      const int chunks = 16 * chunksPerRow;
      for (int j = tid; j < chunks; j += blockDim.x) {
        const int row = j / chunksPerRow, cj = j % chunksPerRow;
        const unsigned short* gsrc = hRead + (size_t)row * HSTRIDE + cj * 8;
        unsigned short*       ldst = h_lds + row * HLDS + cj * 8;
#if ASYNC_LDS
        __builtin_amdgcn_global_load_async_to_lds_b128((AS1 v4i*)gsrc, (AS3 v4i*)ldst, 0, 0);
#else
        *(u32x4*)ldst = *(const u32x4*)gsrc;
#endif
      }
#if ASYNC_LDS
#if __has_builtin(__builtin_amdgcn_s_wait_asynccnt)
      __builtin_amdgcn_s_wait_asynccnt(0);
#else
      asm volatile("s_wait_asynccnt 0" ::: "memory");
#endif
#endif
    }
    __syncthreads();

    // ---- gh tile: WMMA over LDS-resident h ----
    if (wave < ntiles) {                                   // wave-uniform branch
      const int gate  = wave / tpg;
      const int ct    = wave % tpg;
      const int nrow0 = gate * Hl + cs + ct * 16;          // Whh row block (gh column block)
      const int wrow  = nrow0 + (lane & 15);
      const unsigned short* aPtr = h_lds + (lane & 15) * HLDS + ((lane >> 4) * 16);
      const unsigned short* wPtr = Whh   + (size_t)wrow * Hl + ((lane >> 4) * 16);
      f32x8 acc = {};
      for (int kk = 0; kk < Hl; kk += 32) {
        AFrag a, b;
        a.q[0] = *(const u32x4*)(aPtr + kk);               // ds_load_b128
        a.q[1] = *(const u32x4*)(aPtr + kk + 8);
        b.q[0] = *(const u32x4*)(wPtr + kk);
        b.q[1] = *(const u32x4*)(wPtr + kk + 8);
        acc = __builtin_amdgcn_wmma_f32_16x16x32_bf16(false, a.v, false, b.v,
                                                      (short)0, acc, false, false);
      }
      CFrag c; c.v = acc;
      const int n  = lane & 15;
      const int mo = (lane >> 4) * 8;
      const float bb = bhh[nrow0 + n];
      for (int r = 0; r < 8; ++r)
        ghlds[(mo + r) * GHLD + wave * 16 + n] = c.f[r] + bb;   // gh[m][w*16+n] (+bhh)
    }
    __syncthreads();

    // ---- fused gate math on this WG's h-column slice ----
    for (int idx = tid; idx < ncols * 8; idx += blockDim.x) {
      const int b = idx & 7, cI = idx >> 3;
      const int col = cs + cI;
      const int tsub = cI >> 4, n = cI & 15;
      const float ghr = ghlds[b * GHLD + (0 * tpg + tsub) * 16 + n];
      const float ghz = ghlds[b * GHLD + (1 * tpg + tsub) * 16 + n];
      const float ghn = ghlds[b * GHLD + (2 * tpg + tsub) * 16 + n];
      const size_t gib = (size_t)(t * 8 + b) * (3 * Hl);
      const float gir = Gi[gib + col];
      const float giz = Gi[gib + Hl + col];
      const float gin = Gi[gib + 2 * Hl + col];
      const float r  = 1.0f / (1.0f + __expf(-(gir + ghr)));
      const float z  = 1.0f / (1.0f + __expf(-(giz + ghz)));
      const float nn = tanhf(gin + r * ghn);
      const float hprev = hS[(size_t)b * HSTRIDE + col];
      const float hnew  = (1.0f - z) * nn + z * hprev;
      hS[(size_t)b * HSTRIDE + col] = hnew;
      const unsigned short h16 = f2bf(hnew);
      hWrite[(size_t)b * HSTRIDE + col] = h16;
      Xout[(size_t)(t * 8 + b) * Hl + col] = h16;           // layer output (next layer's A)
    }
    grid_barrier(cnt, gen, nWG, s_gen0 + 2 + t);
  }
}

// ---------------------------------------------------------------------------
// Row-wise log_softmax in place; one 256-thread block per row.
// ---------------------------------------------------------------------------
__global__ void log_softmax_rows(float* __restrict__ data, int N) {
  float* row = data + (size_t)blockIdx.x * N;
  __shared__ float red[256];
  const int tid = threadIdx.x;
  float mx = -3.4e38f;
  for (int c = tid; c < N; c += blockDim.x) mx = fmaxf(mx, row[c]);
  red[tid] = mx; __syncthreads();
  for (int s = 128; s > 0; s >>= 1) { if (tid < s) red[tid] = fmaxf(red[tid], red[tid + s]); __syncthreads(); }
  const float m = red[0]; __syncthreads();
  float sum = 0.0f;
  for (int c = tid; c < N; c += blockDim.x) sum += __expf(row[c] - m);
  red[tid] = sum; __syncthreads();
  for (int s = 128; s > 0; s >>= 1) { if (tid < s) red[tid] += red[tid + s]; __syncthreads(); }
  const float ls = __logf(red[0]);
  for (int c = tid; c < N; c += blockDim.x) row[c] = row[c] - m - ls;
}

// ---------------------------------------------------------------------------
// Host orchestration
// d_in: [0]=tokens [1]=E [2..13]=g_params(Wih,Whh,bih,bhh x3) [14..25]=s_params
//       [26]=Wg [27]=bg [28]=Ws [29]=bs
// ---------------------------------------------------------------------------
extern "C" void kernel_launch(void* const* d_in, const int* in_sizes, int n_in,
                              void* d_out, int out_size, void* d_ws, size_t ws_size,
                              hipStream_t stream) {
  (void)in_sizes; (void)n_in; (void)out_size; (void)ws_size;
  const int*   tokens = (const int*)d_in[0];
  const float* E      = (const float*)d_in[1];
  const float *Wih[2][3], *Whh[2][3], *bih[2][3], *bhh[2][3];
  for (int s = 0; s < 2; ++s)
    for (int l = 0; l < 3; ++l) {
      int b0 = 2 + s * 12 + l * 4;
      Wih[s][l] = (const float*)d_in[b0 + 0];
      Whh[s][l] = (const float*)d_in[b0 + 1];
      bih[s][l] = (const float*)d_in[b0 + 2];
      bhh[s][l] = (const float*)d_in[b0 + 3];
    }
  const float* Wg = (const float*)d_in[26];
  const float* bg = (const float*)d_in[27];
  const float* Ws = (const float*)d_in[28];
  const float* bs = (const float*)d_in[29];

  static const int inD[3]  = {512, 1024, 1024};
  static const int hidD[3] = {1024, 1024, 512};

  uint8_t* base = (uint8_t*)d_ws; size_t off = 0;
  auto alloc = [&](size_t bytes) -> void* {
    void* p = base + off; off += (bytes + 255) & ~(size_t)255; return p;
  };

  unsigned short *wihB[2][3], *whhB[2][3];
  for (int s = 0; s < 2; ++s)
    for (int l = 0; l < 3; ++l) {
      wihB[s][l] = (unsigned short*)alloc((size_t)3 * hidD[l] * inD[l]  * 2);
      whhB[s][l] = (unsigned short*)alloc((size_t)3 * hidD[l] * hidD[l] * 2);
    }
  unsigned short* wgB = (unsigned short*)alloc((size_t)50000 * 512 * 2);
  unsigned short* wsB = (unsigned short*)alloc((size_t)25000 * 512 * 2);
  unsigned short* X0  = (unsigned short*)alloc((size_t)MROWS * 512 * 2);
  unsigned short *Xa[2], *Xb[2];
  for (int s = 0; s < 2; ++s) {
    Xa[s] = (unsigned short*)alloc((size_t)MROWS * 1024 * 2);
    Xb[s] = (unsigned short*)alloc((size_t)MROWS * 1024 * 2);
  }
  float* Gi[2];
  for (int s = 0; s < 2; ++s) Gi[s] = (float*)alloc((size_t)MROWS * 3072 * 4);
  float*          hf = (float*)alloc((size_t)2 * 8 * HSTRIDE * 4);
  unsigned short* hb = (unsigned short*)alloc((size_t)2 * 2 * 16 * HSTRIDE * 2);
  int* sync = (int*)alloc(64);

  // --- prep: sync counters, bf16 weights, embedding gather ---
  init_sync<<<1, 64, 0, stream>>>(sync);
  for (int s = 0; s < 2; ++s)
    for (int l = 0; l < 3; ++l) {
      cvt_f32_bf16<<<256, 256, 0, stream>>>(Wih[s][l], wihB[s][l], 3 * hidD[l] * inD[l]);
      cvt_f32_bf16<<<256, 256, 0, stream>>>(Whh[s][l], whhB[s][l], 3 * hidD[l] * hidD[l]);
    }
  cvt_f32_bf16<<<512, 256, 0, stream>>>(Wg, wgB, 50000 * 512);
  cvt_f32_bf16<<<512, 256, 0, stream>>>(Ws, wsB, 25000 * 512);
  gather_embed<<<MROWS, 256, 0, stream>>>(tokens, E, X0);

  // --- GRU stacks: per layer, batched input GEMM then distributed recurrence ---
  for (int l = 0; l < 3; ++l) {
    const unsigned short *in_g, *in_s; unsigned short *out_g, *out_s;
    if (l == 0)      { in_g = X0;    in_s = X0;    out_g = Xa[0]; out_s = Xa[1]; }
    else if (l == 1) { in_g = Xa[0]; in_s = Xa[1]; out_g = Xb[0]; out_s = Xb[1]; }
    else             { in_g = Xb[0]; in_s = Xb[1]; out_g = Xa[0]; out_s = Xa[1]; }
    const int K = inD[l], H = hidD[l], N = 3 * H;
    dim3 gg((N + 511) / 512, MROWS / 16);
    wmma_gemm_bias<<<gg, 256, 0, stream>>>(in_g, wihB[0][l], bih[0][l], Gi[0], N, K);
    wmma_gemm_bias<<<gg, 256, 0, stream>>>(in_s, wihB[1][l], bih[1][l], Gi[1], N, K);
    gru_recurrence<<<dim3(32, 2), 256, 0, stream>>>(Gi[0], Gi[1], whhB[0][l], whhB[1][l],
                                                    bhh[0][l], bhh[1][l], out_g, out_s,
                                                    hf, hb, sync, H);
  }

  // --- heads + log_softmax (final activations live in Xa, stride 512) ---
  float* outG = (float*)d_out;
  float* outS = (float*)d_out + (size_t)MROWS * 50000;
  wmma_gemm_bias<<<dim3((50000 + 511) / 512, MROWS / 16), 256, 0, stream>>>(Xa[0], wgB, bg, outG, 50000, 512);
  wmma_gemm_bias<<<dim3((25000 + 511) / 512, MROWS / 16), 256, 0, stream>>>(Xa[1], wsB, bs, outS, 25000, 512);
  log_softmax_rows<<<MROWS, 256, 0, stream>>>(outG, 50000);
  log_softmax_rows<<<MROWS, 256, 0, stream>>>(outS, 25000);
}
